// ProxyStudentTAttention_78245714199344
// MI455X (gfx1250) — matrix operand order repaired and checked by reading, not compiled, not run on hardware
//
#include <hip/hip_runtime.h>
#include <hip/hip_bf16.h>

typedef __attribute__((ext_vector_type(16))) _Float16 v16h;
typedef __attribute__((ext_vector_type(8)))  _Float16 v8h;
typedef __attribute__((ext_vector_type(8)))  float    v8f;

#define NB 4
#define NT 1024
#define NC 1024
#define NH 16
#define NDH 64

__device__ __forceinline__ float fast_tanh(float x) {
  // tanh(x) = 1 - 2/(1+e^{2x}); exact at tails, v_exp + v_rcp, no IEEE divide
  float e = __expf(2.0f * x);
  return 1.0f - 2.0f * __builtin_amdgcn_rcpf(1.0f + e);
}

// ---------------------------------------------------------------- elementwise
__global__ __launch_bounds__(256) void tanh_k(const float* __restrict__ in,
                                              float* __restrict__ out, int n) {
  int i = blockIdx.x * 256 + threadIdx.x;
  if (i < n) out[i] = fast_tanh(in[i]);
}

__global__ __launch_bounds__(256) void cvt_f16_k(const float* __restrict__ in,
                                                 _Float16* __restrict__ out, int n) {
  int i = blockIdx.x * 256 + threadIdx.x;
  if (i < n) out[i] = (_Float16)in[i];
}

// ---------------------------------------------------------------- L1-cdist projection
// P[i][j] = (0.5 - mean_d |xa[i][d] - wa[j][d]|) * gain[j]
// MODE 0: write f32 at [bh][t][dh]   (for q,k)
// MODE 1: write f16 at [bh][dh][t]   (transposed v, WMMA-B friendly)
template <int MODE>
__global__ __launch_bounds__(256) void proj_l1_k(const float* __restrict__ xa,
                                                 const float* __restrict__ wa,
                                                 const float* __restrict__ gain,
                                                 float* __restrict__ outQK,
                                                 _Float16* __restrict__ outVT) {
  __shared__ float xs[64][34];   // K-chunk 32, pad 34: b64 reads, <=2-way conflicts
  __shared__ float ws[64][34];
  const int t  = threadIdx.x;
  const int tx = t & 15, ty = t >> 4;
  const int i0 = blockIdx.y * 64, j0 = blockIdx.x * 64;

  float acc[4][4] = {};
  for (int d0 = 0; d0 < NC; d0 += 32) {
#pragma unroll
    for (int rep = 0; rep < 2; ++rep) {
      int idx = t + rep * 256;            // 512 float4s per 64x32 tile
      int rr = idx >> 3, cc = (idx & 7) * 4;
      float4 xv = *(const float4*)(xa + (size_t)(i0 + rr) * NC + d0 + cc);
      float4 wv = *(const float4*)(wa + (size_t)(j0 + rr) * NC + d0 + cc);
      xs[rr][cc + 0] = xv.x; xs[rr][cc + 1] = xv.y; xs[rr][cc + 2] = xv.z; xs[rr][cc + 3] = xv.w;
      ws[rr][cc + 0] = wv.x; ws[rr][cc + 1] = wv.y; ws[rr][cc + 2] = wv.z; ws[rr][cc + 3] = wv.w;
    }
    __syncthreads();
#pragma unroll
    for (int dd = 0; dd < 16; ++dd) {     // 2 features per step (float2 LDS reads)
      float2 a0 = *(const float2*)&xs[ty * 4 + 0][dd * 2];
      float2 a1 = *(const float2*)&xs[ty * 4 + 1][dd * 2];
      float2 a2 = *(const float2*)&xs[ty * 4 + 2][dd * 2];
      float2 a3 = *(const float2*)&xs[ty * 4 + 3][dd * 2];
      float2 b0 = *(const float2*)&ws[tx * 4 + 0][dd * 2];
      float2 b1 = *(const float2*)&ws[tx * 4 + 1][dd * 2];
      float2 b2 = *(const float2*)&ws[tx * 4 + 2][dd * 2];
      float2 b3 = *(const float2*)&ws[tx * 4 + 3][dd * 2];
      acc[0][0] += fabsf(a0.x - b0.x) + fabsf(a0.y - b0.y);
      acc[0][1] += fabsf(a0.x - b1.x) + fabsf(a0.y - b1.y);
      acc[0][2] += fabsf(a0.x - b2.x) + fabsf(a0.y - b2.y);
      acc[0][3] += fabsf(a0.x - b3.x) + fabsf(a0.y - b3.y);
      acc[1][0] += fabsf(a1.x - b0.x) + fabsf(a1.y - b0.y);
      acc[1][1] += fabsf(a1.x - b1.x) + fabsf(a1.y - b1.y);
      acc[1][2] += fabsf(a1.x - b2.x) + fabsf(a1.y - b2.y);
      acc[1][3] += fabsf(a1.x - b3.x) + fabsf(a1.y - b3.y);
      acc[2][0] += fabsf(a2.x - b0.x) + fabsf(a2.y - b0.y);
      acc[2][1] += fabsf(a2.x - b1.x) + fabsf(a2.y - b1.y);
      acc[2][2] += fabsf(a2.x - b2.x) + fabsf(a2.y - b2.y);
      acc[2][3] += fabsf(a2.x - b3.x) + fabsf(a2.y - b3.y);
      acc[3][0] += fabsf(a3.x - b0.x) + fabsf(a3.y - b0.y);
      acc[3][1] += fabsf(a3.x - b1.x) + fabsf(a3.y - b1.y);
      acc[3][2] += fabsf(a3.x - b2.x) + fabsf(a3.y - b2.y);
      acc[3][3] += fabsf(a3.x - b3.x) + fabsf(a3.y - b3.y);
    }
    __syncthreads();
  }
#pragma unroll
  for (int ii = 0; ii < 4; ++ii)
#pragma unroll
    for (int jj = 0; jj < 4; ++jj) {
      int i = i0 + ty * 4 + ii, j = j0 + tx * 4 + jj;
      float val = (0.5f - acc[ii][jj] * (1.0f / 1024.0f)) * gain[j];
      int b = i >> 10, tt = i & 1023, h = j >> 6, dh = j & 63;
      int bh = b * NH + h;
      if (MODE == 0)
        outQK[((size_t)bh * NT + tt) * NDH + dh] = val;
      else
        outVT[((size_t)bh * NDH + dh) * NT + tt] = (_Float16)val;
    }
}

// per-lane async 16B copy global -> LDS (ASYNCcnt-tracked, no VGPR round trip)
__device__ __forceinline__ void async_copy_b128(void* lds_ptr, const void* gptr) {
  unsigned lds_off = (unsigned)(size_t)lds_ptr;          // low 32 bits = LDS offset
  unsigned long long ga = (unsigned long long)(size_t)gptr;
  asm volatile("global_load_async_to_lds_b128 %0, %1, off"
               :: "v"(lds_off), "v"(ga) : "memory");
}
__device__ __forceinline__ void wait_async0() {
  asm volatile("s_wait_asynccnt 0x0" ::: "memory");
}

// ---------------------------------------------------------------- fused Student-T attention
// one workgroup = 64 queries of one (b,h); loops over 16 key tiles of 64.
__global__ __launch_bounds__(256) void attn_k(const float* __restrict__ qb,
                                              const float* __restrict__ kb,
                                              const _Float16* __restrict__ vT,
                                              const float* __restrict__ gamma,
                                              const float* __restrict__ rho,
                                              _Float16* __restrict__ attnH) {
  __shared__ float    qsh[64][64];
  __shared__ float    ksh[64][64];
  __shared__ _Float16 wt[64][72];    // stride 72 halves: 16B-aligned frags, low-conflict stores
  __shared__ float    rowsum[64];

  const int t    = threadIdx.x;
  const int lane = t & 31, wv = t >> 5;
  const int bh   = blockIdx.y;
  const int h    = bh & (NH - 1), b = bh >> 4;
  const int q0   = blockIdx.x * 64;

  const float g = log1pf(__expf(gamma[h])) + 1e-4f;
  const float r = log1pf(__expf(rho[h])) + 1e-4f;

  const float*    qbase = qb + ((size_t)bh * NT + q0) * NDH;
  const float*    kbase = kb + (size_t)bh * NT * NDH;
  const _Float16* vbase = vT + (size_t)bh * NDH * NT;

  // async-stage the 64x64 q tile
#pragma unroll
  for (int rep = 0; rep < 4; ++rep) {
    int idx = t + rep * 256;                 // 1024 float4s
    int rr = idx >> 4, cc = (idx & 15) * 4;
    async_copy_b128(&qsh[rr][cc], qbase + (size_t)rr * NDH + cc);
  }
  if (t < 64) rowsum[t] = 0.0f;
  wait_async0();
  __syncthreads();

  const int qloc = t & 63, kgrp = t >> 6;
  float4 qreg[16];
#pragma unroll
  for (int d4 = 0; d4 < 16; ++d4) qreg[d4] = *(const float4*)&qsh[qloc][d4 * 4];

  const int rsub = lane & 15, half = lane >> 4;
  const int qt = wv >> 1;
  const int nt0 = (wv & 1) * 2, nt1 = nt0 + 1;
  v8f acc0 = {}, acc1 = {};

  for (int kt = 0; kt < 16; ++kt) {
    // async-stage the key tile (safe: all reads of ksh finished before last barrier)
#pragma unroll
    for (int rep = 0; rep < 4; ++rep) {
      int idx = t + rep * 256;
      int rr = idx >> 4, cc = (idx & 15) * 4;
      async_copy_b128(&ksh[rr][cc], kbase + (size_t)(kt * 64 + rr) * NDH + cc);
    }
    {  // prefetch the next tile into L2 (global_prefetch_b8)
      int ktn = (kt + 1 < 16) ? kt + 1 : 15;
      __builtin_prefetch(kbase + (size_t)ktn * 64 * NDH + t * 16, 0, 0);
    }
    wait_async0();
    __syncthreads();

    float psum = 0.0f;
#pragma unroll
    for (int kk = 0; kk < 16; ++kk) {
      int k = kgrp * 16 + kk;
      const float4* krow = (const float4*)&ksh[k][0];
      float s = 0.0f;
#pragma unroll
      for (int d4 = 0; d4 < 16; ++d4) {
        float4 kv = krow[d4];
        float4 qv = qreg[d4];
        s += fabsf(qv.x - kv.x) + fabsf(qv.y - kv.y) +
             fabsf(qv.z - kv.z) + fabsf(qv.w - kv.w);
      }
      // w = (1 + g*dist)^(-r), dist = s/64; base >= 1 so log is safe
      float wval = __expf(-r * __logf(fmaf(g, s * (1.0f / 64.0f), 1.0f)));
      wt[qloc][k] = (_Float16)wval;
      psum += wval;
    }
    atomicAdd(&rowsum[qloc], psum);   // ds_add_f32
    __syncthreads();

    // acc += W_tile(64x64) x Vt_tile(64k x 64dh), two 16x16 C tiles per wave
#pragma unroll
    for (int kc = 0; kc < 2; ++kc) {
      const _Float16* arow = &wt[qt * 16 + rsub][kc * 32];
      v8h alo = *(const v8h*)(arow + 16 * half);
      v8h ahi = *(const v8h*)(arow + 16 * half + 8);
      v16h afr = __builtin_shufflevector(alo, ahi, 0, 1, 2, 3, 4, 5, 6, 7,
                                         8, 9, 10, 11, 12, 13, 14, 15);
      int kglob = kt * 64 + kc * 32 + 16 * half;
      v16h b0 = *(const v16h*)(vbase + (size_t)(nt0 * 16 + rsub) * NT + kglob);
      v16h b1 = *(const v16h*)(vbase + (size_t)(nt1 * 16 + rsub) * NT + kglob);
      acc0 = __builtin_amdgcn_wmma_f32_16x16x32_f16(false, afr, false, b0,
                                                    (short)0, acc0, false, false);
      acc1 = __builtin_amdgcn_wmma_f32_16x16x32_f16(false, afr, false, b1,
                                                    (short)0, acc1, false, false);
    }
    __syncthreads();
  }

  // normalize by (rowsum + 1e-6) via v_rcp_f32 and store f16 [b][t][h*64+n]
#pragma unroll
  for (int i = 0; i < 8; ++i) {
    int m = qt * 16 + half * 8 + i;
    float inv = __builtin_amdgcn_rcpf(rowsum[m] + 1e-6f);
    size_t base = ((size_t)b * NT + q0 + m) * NC + h * NDH;
    attnH[base + nt0 * 16 + rsub] = (_Float16)(acc0[i] * inv);
    attnH[base + nt1 * 16 + rsub] = (_Float16)(acc1[i] * inv);
  }
}

// ---------------------------------------------------------------- output projection GEMM
// out[i][j] = sum_d A[i][d]*W[j][d] + bias[j]; A,W row-major f16. Pure WMMA.
__global__ __launch_bounds__(256) void out_gemm_k(const _Float16* __restrict__ A,
                                                  const _Float16* __restrict__ W,
                                                  const float* __restrict__ bias,
                                                  float* __restrict__ out) {
  const int t = threadIdx.x, lane = t & 31, wv = t >> 5;
  const int rsub = lane & 15, half = lane >> 4;
  const int i0 = blockIdx.y * 64, j0 = blockIdx.x * 64;
  const int qt = wv >> 1, nt0 = (wv & 1) * 2, nt1 = nt0 + 1;
  v8f acc0 = {}, acc1 = {};
  const _Float16* arow = A + (size_t)(i0 + qt * 16 + rsub) * NC;
  const _Float16* w0   = W + (size_t)(j0 + nt0 * 16 + rsub) * NC;
  const _Float16* w1   = W + (size_t)(j0 + nt1 * 16 + rsub) * NC;
  for (int k0 = 0; k0 < NC; k0 += 32) {
    v8h alo = *(const v8h*)(arow + k0 + 16 * half);
    v8h ahi = *(const v8h*)(arow + k0 + 16 * half + 8);
    v16h afr = __builtin_shufflevector(alo, ahi, 0, 1, 2, 3, 4, 5, 6, 7,
                                       8, 9, 10, 11, 12, 13, 14, 15);
    v16h b0 = *(const v16h*)(w0 + k0 + 16 * half);
    v16h b1 = *(const v16h*)(w1 + k0 + 16 * half);
    acc0 = __builtin_amdgcn_wmma_f32_16x16x32_f16(false, afr, false, b0,
                                                  (short)0, acc0, false, false);
    acc1 = __builtin_amdgcn_wmma_f32_16x16x32_f16(false, afr, false, b1,
                                                  (short)0, acc1, false, false);
  }
#pragma unroll
  for (int i = 0; i < 8; ++i) {
    int m = i0 + qt * 16 + half * 8 + i;
    int n0 = j0 + nt0 * 16 + rsub, n1 = j0 + nt1 * 16 + rsub;
    out[(size_t)m * NC + n0] = acc0[i] + bias[n0];
    out[(size_t)m * NC + n1] = acc1[i] + bias[n1];
  }
}

// ---------------------------------------------------------------- launch
extern "C" void kernel_launch(void* const* d_in, const int* in_sizes, int n_in,
                              void* d_out, int out_size, void* d_ws, size_t ws_size,
                              hipStream_t stream) {
  const float* x     = (const float*)d_in[0];
  const float* wq_w  = (const float*)d_in[1];
  const float* wq_g  = (const float*)d_in[2];
  const float* wk_w  = (const float*)d_in[3];
  const float* wk_g  = (const float*)d_in[4];
  const float* wv_w  = (const float*)d_in[5];
  const float* wv_g  = (const float*)d_in[6];
  const float* wo_w  = (const float*)d_in[7];
  const float* wo_b  = (const float*)d_in[8];
  const float* gamma = (const float*)d_in[9];
  const float* rho   = (const float*)d_in[10];
  float* out = (float*)d_out;

  char* ws = (char*)d_ws;
  float*    xa = (float*)(ws);                          // 16 MB: tanh(x) [4096,1024]
  float*    wa = (float*)(ws + ((size_t)16 << 20));     //  4 MB: tanh(W) (reused q/k/v)
  float*    qb = (float*)(ws + ((size_t)20 << 20));     // 16 MB: q [bh,t,dh] f32
  float*    kb = (float*)(ws + ((size_t)36 << 20));     // 16 MB: k [bh,t,dh] f32
  _Float16* vT = (_Float16*)(ws + ((size_t)52 << 20));  //  8 MB: v [bh,dh,t] f16
  _Float16* aH = (_Float16*)(ws + ((size_t)60 << 20));  //  8 MB: attn out [b,t,c] f16
  _Float16* wH = (_Float16*)(ws + ((size_t)68 << 20));  //  2 MB: wo f16

  dim3 blk(256);
  dim3 pg(NC / 64, (NB * NT) / 64);   // (16, 64)
  const int nX = NB * NT * NC;        // 4M
  const int nW = NC * NC;             // 1M

  tanh_k<<<nX / 256, blk, 0, stream>>>(x, xa, nX);

  tanh_k<<<nW / 256, blk, 0, stream>>>(wq_w, wa, nW);
  proj_l1_k<0><<<pg, blk, 0, stream>>>(xa, wa, wq_g, qb, nullptr);

  tanh_k<<<nW / 256, blk, 0, stream>>>(wk_w, wa, nW);
  proj_l1_k<0><<<pg, blk, 0, stream>>>(xa, wa, wk_g, kb, nullptr);

  tanh_k<<<nW / 256, blk, 0, stream>>>(wv_w, wa, nW);
  proj_l1_k<1><<<pg, blk, 0, stream>>>(xa, wa, wv_g, nullptr, vT);

  cvt_f16_k<<<nW / 256, blk, 0, stream>>>(wo_w, wH, nW);

  attn_k<<<dim3(NT / 64, NB * NH), blk, 0, stream>>>(qb, kb, vT, gamma, rho, aH);

  out_gemm_k<<<dim3(NC / 64, (NB * NT) / 64), blk, 0, stream>>>(aH, wH, wo_b, out);
}
